// PSMPairPlainEncoderLayer_19567871001120
// MI455X (gfx1250) — matrix operand order repaired and checked by reading, not compiled, not run on hardware
//
#include <hip/hip_runtime.h>
#include <cstdint>
#include <cstddef>

// ---------------------------------------------------------------------------
// PSMPairPlainEncoderLayer for MI455X (gfx1250, wave32, WMMA f32_16x16x32_f16)
// v2: 2x2 register-blocked WMMA GEMM (+prefetch), 1x2-blocked flash attention
// ---------------------------------------------------------------------------
static constexpr int Ld  = 512;
static constexpr int Bd  = 4;
static constexpr int Dd  = 1024;
static constexpr int Hd  = 16;
static constexpr int DHd = 64;
static constexpr int FFd = 4096;
static constexpr int HPd = 32;
static constexpr int Kd  = 256;
static constexpr int LKd = Ld + Kd;      // 768
static constexpr int LB  = Ld * Bd;      // 2048 rows
static constexpr int LKB = LKd * Bd;     // 3072 rows

typedef __attribute__((ext_vector_type(16))) _Float16 v16h;
typedef __attribute__((ext_vector_type(8)))  float    v8f;

// ---- WMMA fragment loaders (CDNA5 ISA 7.12.2 wave32 layouts) --------------
// A (16x32 f16): lanes m and m+16 hold row M=m; lane-half selects K groups:
//   lanes 0-15 : v0..v3 K=0..7,  v4..v7 K=16..23
//   lanes 16-31: v0..v3 K=8..15, v4..v7 K=24..31
__device__ __forceinline__ v16h load_a_frag(const _Float16* __restrict__ A,
                                            int lda, int k0, int lane) {
  union { v16h h; uint32_t u[8]; } r;
  const int m  = lane & 15;
  const int hf = lane >> 4;
  const uint32_t* row = (const uint32_t*)(A + (size_t)m * lda + k0);
  const int o = 4 * hf;
#pragma unroll
  for (int i = 0; i < 4; ++i) r.u[i]     = row[o + i];       // K = 8*hf + 2i
#pragma unroll
  for (int i = 0; i < 4; ++i) r.u[4 + i] = row[8 + o + i];   // K = 16 + 8*hf + 2i
  return r.h;
}

// B (32x16 f16) given as Bt: row n of Bt = column n of B, contiguous in K.
//   lanes n and n+16 hold column N=n; same K order as A: lanes 0-15: K=0..7,16..23; lanes 16-31: K=8..15,24..31
__device__ __forceinline__ v16h load_b_frag(const _Float16* __restrict__ Bt,
                                            int ldb, int k0, int lane) {
  union { v16h h; uint32_t u[8]; } r;
  const int n  = lane & 15;
  const int hf = lane >> 4;
  const uint32_t* row = (const uint32_t*)(Bt + (size_t)n * ldb + k0);
  const int o = 4 * hf;
#pragma unroll
  for (int i = 0; i < 4; ++i) r.u[i]     = row[o + i];       // K = 8*hf + 2i
#pragma unroll
  for (int i = 0; i < 4; ++i) r.u[4 + i] = row[8 + o + i];   // K = 16 + 8*hf + 2i
  return r.h;
}

__device__ __forceinline__ v8f wmma_f16(v16h a, v16h b, v8f c) {
  return __builtin_amdgcn_wmma_f32_16x16x32_f16(false, a, false, b,
                                                (short)0, c, false, false);
}

// ---- generic WMMA GEMM: C(MxN) = act(A(MxK) * Bt(NxK)^T) [+ residual] -----
// act: 0=none, 1=exact GELU, 2=SiLU.
// Block = 256 thr = 8 waves (2x4); each wave owns a 32x32 tile (2x2 WMMA).
__global__ void gemm_wmma_kernel(const _Float16* __restrict__ A,
                                 const _Float16* __restrict__ Bt,
                                 const float* __restrict__ residual,
                                 float* __restrict__ Cf,
                                 _Float16* __restrict__ Ch,
                                 int M, int N, int Kdim, int act) {
  const int tid  = threadIdx.x;
  const int wave = tid >> 5;
  const int lane = tid & 31;
  const int m    = lane & 15;
  const int hf   = lane >> 4;
  const int tm   = wave >> 2;           // 0..1
  const int tn   = wave & 3;            // 0..3
  const int rowBase = blockIdx.y * 64 + tm * 32;
  const int colBase = blockIdx.x * 128 + tn * 32;
  if (rowBase >= M || colBase >= N) return;   // wave-uniform

  v8f c00 = {}, c01 = {}, c10 = {}, c11 = {};
  const _Float16* A0 = A  + (size_t)rowBase * Kdim;
  const _Float16* A1 = A0 + (size_t)16 * Kdim;
  const _Float16* B0 = Bt + (size_t)colBase * Kdim;
  const _Float16* B1 = B0 + (size_t)16 * Kdim;

  for (int k0 = 0; k0 < Kdim; k0 += 32) {
    if (k0 + 128 < Kdim) {   // stream next chunks into cache (global_prefetch_b8)
      __builtin_prefetch(A0 + (size_t)m * Kdim + k0 + 128, 0, 3);
      __builtin_prefetch(A1 + (size_t)m * Kdim + k0 + 128, 0, 3);
      __builtin_prefetch(B0 + (size_t)m * Kdim + k0 + 128, 0, 3);
      __builtin_prefetch(B1 + (size_t)m * Kdim + k0 + 128, 0, 3);
    }
    v16h a0 = load_a_frag(A0, Kdim, k0, lane);
    v16h a1 = load_a_frag(A1, Kdim, k0, lane);
    v16h b0 = load_b_frag(B0, Kdim, k0, lane);
    v16h b1 = load_b_frag(B1, Kdim, k0, lane);
    c00 = wmma_f16(a0, b0, c00);
    c01 = wmma_f16(a0, b1, c01);
    c10 = wmma_f16(a1, b0, c10);
    c11 = wmma_f16(a1, b1, c11);
  }

  auto store_tile = [&](v8f acc, int rB, int cB) {
    const int col = cB + m;
#pragma unroll
    for (int r = 0; r < 8; ++r) {
      const int row = rB + r + 8 * hf;
      const size_t idx = (size_t)row * N + col;
      float v = acc[r];
      if (act == 1)      v = 0.5f * v * (1.0f + erff(v * 0.70710678118654752f));
      else if (act == 2) v = v / (1.0f + __expf(-v));
      if (residual) v += residual[idx];
      if (Cf) Cf[idx] = v;
      if (Ch) Ch[idx] = (_Float16)v;
    }
  };
  store_tile(c00, rowBase,      colBase);
  store_tile(c01, rowBase,      colBase + 16);
  store_tile(c10, rowBase + 16, colBase);
  store_tile(c11, rowBase + 16, colBase + 16);
}

// ---- LayerNorm (per row of D), write f16 ----------------------------------
__global__ void ln_f16_kernel(const float* __restrict__ x,
                              const float* __restrict__ g,
                              const float* __restrict__ be,
                              _Float16* __restrict__ outh, int D) {
  __shared__ float red[256];
  const int row = blockIdx.x, tid = threadIdx.x;
  const float* xr = x + (size_t)row * D;
  float s = 0.f;
  for (int i = tid; i < D; i += 256) s += xr[i];
  red[tid] = s; __syncthreads();
  for (int st = 128; st; st >>= 1) { if (tid < st) red[tid] += red[tid + st]; __syncthreads(); }
  const float mean = red[0] / D;
  __syncthreads();
  float s2 = 0.f;
  for (int i = tid; i < D; i += 256) { float d = xr[i] - mean; s2 += d * d; }
  red[tid] = s2; __syncthreads();
  for (int st = 128; st; st >>= 1) { if (tid < st) red[tid] += red[tid + st]; __syncthreads(); }
  const float inv = rsqrtf(red[0] / D + 1e-5f);
  for (int i = tid; i < D; i += 256) {
    float v = (xr[i] - mean) * inv * g[i] + be[i];
    outh[(size_t)row * D + i] = (_Float16)v;
  }
}

// ---- weight cast + transpose: Wt[n*K+k] = (f16)W[k*N+n] -------------------
__global__ void cvt_wT_kernel(const float* __restrict__ W,
                              _Float16* __restrict__ Wt, int K, int N) {
  const int idx = blockIdx.x * 256 + threadIdx.x;
  if (idx >= K * N) return;
  const int n = idx % N, k = idx / N;
  Wt[(size_t)n * K + k] = (_Float16)W[idx];
}

// ---- build kv_src (L+K,B,D) f16 from h16 via outcell gather ---------------
__global__ void kv_gather_kernel(const _Float16* __restrict__ h16,
                                 const int* __restrict__ oc,
                                 _Float16* __restrict__ kv16) {
  const int idx = blockIdx.x * 256 + threadIdx.x;
  if (idx >= LKB * Dd) return;
  const int d = idx & (Dd - 1);
  const int row = idx >> 10;
  const int lk = row >> 2, b = row & 3;
  const int src_row = (lk < Ld) ? row : (oc[b * Kd + (lk - Ld)] * Bd + b);
  kv16[idx] = h16[(size_t)src_row * Dd + d];
}

// ---- RoPE Q: (L,B,H*DH) f32 -> (B,H,L,DH) f16 -----------------------------
__global__ void rope_q_kernel(const float* __restrict__ qf,
                              const int* __restrict__ pos_ids,
                              _Float16* __restrict__ q16) {
  const int idx = blockIdx.x * 256 + threadIdx.x;   // L*B*H*32
  if (idx >= Ld * Bd * Hd * 32) return;
  const int d2 = idx & 31;
  const int h  = (idx >> 5) & 15;
  const int b  = (idx >> 9) & 3;
  const int l  = idx >> 11;
  const float pos = (float)pos_ids[b * Ld + l];
  const float ang = pos * __expf(-(float)d2 * 0.28782313662425572f); // ln(1e4)/32
  const float s = __sinf(ang), c = __cosf(ang);
  const size_t src = (size_t)(l * Bd + b) * Dd + h * DHd;
  const float x1 = qf[src + d2], x2 = qf[src + 32 + d2];
  const size_t dst = (size_t)((b * Hd + h) * Ld + l) * DHd;
  q16[dst + d2]      = (_Float16)(x1 * c - x2 * s);
  q16[dst + 32 + d2] = (_Float16)(x1 * s + x2 * c);
}

// ---- RoPE K: (L+K,B,H*DH) f32 -> (B,H,LK,DH) f16 (gathered positions) -----
__global__ void rope_k_kernel(const float* __restrict__ kf,
                              const int* __restrict__ pos_ids,
                              const int* __restrict__ oc,
                              _Float16* __restrict__ k16) {
  const int idx = blockIdx.x * 256 + threadIdx.x;   // LK*B*H*32
  if (idx >= LKd * Bd * Hd * 32) return;
  const int d2 = idx & 31;
  const int h  = (idx >> 5) & 15;
  const int b  = (idx >> 9) & 3;
  const int lk = idx >> 11;
  const int pl = (lk < Ld) ? lk : oc[b * Kd + (lk - Ld)];
  const float pos = (float)pos_ids[b * Ld + pl];
  const float ang = pos * __expf(-(float)d2 * 0.28782313662425572f);
  const float s = __sinf(ang), c = __cosf(ang);
  const size_t src = (size_t)(lk * Bd + b) * Dd + h * DHd;
  const float x1 = kf[src + d2], x2 = kf[src + 32 + d2];
  const size_t dst = (size_t)((b * Hd + h) * LKd + lk) * DHd;
  k16[dst + d2]      = (_Float16)(x1 * c - x2 * s);
  k16[dst + 32 + d2] = (_Float16)(x1 * s + x2 * c);
}

// ---- V transpose: (L+K,B,H*DH) f32 -> (B,H,DH,LK) f16 ---------------------
__global__ void vtrans_kernel(const float* __restrict__ vf,
                              _Float16* __restrict__ vt16) {
  const int idx = blockIdx.x * 256 + threadIdx.x;   // LK*B*H*DH
  if (idx >= LKB * Dd) return;
  const int d  = idx & 63;
  const int h  = (idx >> 6) & 15;
  const int b  = (idx >> 10) & 3;
  const int lk = idx >> 12;
  const float v = vf[(size_t)(lk * Bd + b) * Dd + h * DHd + d];
  vt16[(size_t)((b * Hd + h) * DHd + d) * LKd + lk] = (_Float16)v;
}

// ---- fused flash attention: per (b,h), 64 query rows per block ------------
// Wave w owns row-tile tm=w>>1 and key/value tiles tn0,tn0+1 (shared Q/P frag).
__global__ void attn_kernel(const _Float16* __restrict__ q16,
                            const _Float16* __restrict__ k16,
                            const _Float16* __restrict__ vt16,
                            const unsigned char* __restrict__ pad_mask,
                            const unsigned char* __restrict__ exp_mask,
                            _Float16* __restrict__ attn16) {
  __shared__ float    Ss[64 * 64];
  __shared__ _Float16 Ps[64 * 64];
  __shared__ float mrow[64], lrow[64], srow[64];

  const int bh = blockIdx.y;
  const int b = bh / Hd, h = bh % Hd;
  const int qbase = blockIdx.x * 64;
  const _Float16* Q  = q16  + (size_t)bh * Ld  * DHd;  // (L ,DH)
  const _Float16* Km = k16  + (size_t)bh * LKd * DHd;  // (LK,DH)  == Bt for S
  const _Float16* Vt = vt16 + (size_t)bh * DHd * LKd;  // (DH,LK)  == Bt for PV

  const int tid = threadIdx.x, wave = tid >> 5, lane = tid & 31;
  const int m = lane & 15, hf = lane >> 4;
  const int tm  = wave >> 1;          // 0..3 (row tile)
  const int tn0 = (wave & 1) * 2;     // 0 or 2 (first of two col tiles)

  v8f oacc0 = {}, oacc1 = {};
  if (tid < 64) { mrow[tid] = -3.0e38f; lrow[tid] = 0.f; }
  __syncthreads();

  for (int c0 = 0; c0 < LKd; c0 += 64) {
    // ---- phase 1: S = (Q Kt)/8 + mask (shared Q-frag across two K tiles)
    {
      v8f s0 = {}, s1 = {};
      const _Float16* Ab  = Q  + (size_t)(qbase + tm * 16) * DHd;
      const _Float16* Bb0 = Km + (size_t)(c0 + tn0 * 16) * DHd;
      const _Float16* Bb1 = Bb0 + (size_t)16 * DHd;
#pragma unroll
      for (int k0 = 0; k0 < DHd; k0 += 32) {
        v16h a  = load_a_frag(Ab,  DHd, k0, lane);
        v16h b0 = load_b_frag(Bb0, DHd, k0, lane);
        v16h b1 = load_b_frag(Bb1, DHd, k0, lane);
        s0 = wmma_f16(a, b0, s0);
        s1 = wmma_f16(a, b1, s1);
      }
#pragma unroll
      for (int j = 0; j < 2; ++j) {
        const v8f acc = j ? s1 : s0;
        const int col = (tn0 + j) * 16 + m;
        const int key = c0 + col;
        const bool msk = (key < Ld) ? (pad_mask[b * Ld + key] != 0)
                                    : (exp_mask[b * Kd + key - Ld] != 0);
        const float addv = msk ? -1.0e9f : 0.f;
#pragma unroll
        for (int r = 0; r < 8; ++r)
          Ss[(tm * 16 + r + 8 * hf) * 64 + col] = acc[r] * 0.125f + addv;
      }
    }
    __syncthreads();

    // ---- phase 2: online softmax over 64-key chunk (one thread per row)
    if (tid < 64) {
      const int r = tid;
      const float mo = mrow[r];
      float nm = mo;
      for (int c = 0; c < 64; ++c) nm = fmaxf(nm, Ss[r * 64 + c]);
      const float sc = __expf(mo - nm);
      float sum = 0.f;
      for (int c = 0; c < 64; ++c) {
        const float p = __expf(Ss[r * 64 + c] - nm);
        sum += p;
        Ps[r * 64 + c] = (_Float16)p;
      }
      lrow[r] = lrow[r] * sc + sum;
      mrow[r] = nm;
      srow[r] = sc;
    }
    __syncthreads();

    // ---- phase 3: rescale O; accumulate P*V (shared P-frag, two V tiles)
    {
#pragma unroll
      for (int r = 0; r < 8; ++r) {
        const float sc = srow[tm * 16 + r + 8 * hf];
        oacc0[r] *= sc;
        oacc1[r] *= sc;
      }
      const _Float16* Ab  = (const _Float16*)&Ps[tm * 16 * 64];
      const _Float16* Bb0 = Vt + (size_t)(tn0 * 16) * LKd + c0;
      const _Float16* Bb1 = Bb0 + (size_t)16 * LKd;
#pragma unroll
      for (int k0 = 0; k0 < 64; k0 += 32) {
        v16h a  = load_a_frag(Ab,  64,  k0, lane);
        v16h b0 = load_b_frag(Bb0, LKd, k0, lane);
        v16h b1 = load_b_frag(Bb1, LKd, k0, lane);
        oacc0 = wmma_f16(a, b0, oacc0);
        oacc1 = wmma_f16(a, b1, oacc1);
      }
    }
    __syncthreads();
  }

  // ---- epilogue: O / l -> attn16 laid out (L,B,D), D = H*DH
#pragma unroll
  for (int j = 0; j < 2; ++j) {
    const v8f acc = j ? oacc1 : oacc0;
    const int col = (tn0 + j) * 16 + m;   // d within head
#pragma unroll
    for (int r = 0; r < 8; ++r) {
      const int lr = tm * 16 + r + 8 * hf;
      const float o = acc[r] / lrow[lr];
      const int l = qbase + lr;
      attn16[(size_t)(l * Bd + b) * Dd + h * DHd + col] = (_Float16)o;
    }
  }
}

// ---- x_p_j build: (LK,B,HP) = [x_p_i ; gather(x_p_i)] ---------------------
__global__ void xpj_kernel(const float* __restrict__ xpi,
                           const int* __restrict__ oc,
                           float* __restrict__ xpj) {
  const int idx = blockIdx.x * 256 + threadIdx.x;   // LK*B*HP
  if (idx >= LKd * Bd * HPd) return;
  const int h = idx & 31;
  const int b = (idx >> 5) & 3;
  const int k = idx >> 7;
  xpj[idx] = (k < Ld) ? xpi[idx]
                      : xpi[(size_t)(oc[b * Kd + (k - Ld)] * Bd + b) * HPd + h];
}

// ---- pair update: out[l,k,b,h] = x_pair + xpi[l,b,h]*xpj[k,b,h] -----------
__global__ void pair_outer_kernel(const float* __restrict__ x_pair,
                                  const float* __restrict__ xpi,
                                  const float* __restrict__ xpj,
                                  float* __restrict__ out, size_t n) {
  const size_t idx = (size_t)blockIdx.x * 256 + threadIdx.x;
  if (idx >= n) return;
  const int h = (int)(idx & 31);
  const int b = (int)((idx >> 5) & 3);
  const size_t rest = idx >> 7;
  const int k = (int)(rest % LKd);
  const int l = (int)(rest / LKd);
  out[idx] = x_pair[idx] + xpi[(size_t)(l * Bd + b) * HPd + h]
                         * xpj[(size_t)(k * Bd + b) * HPd + h];
}

// ---------------------------------------------------------------------------
extern "C" void kernel_launch(void* const* d_in, const int* in_sizes, int n_in,
                              void* d_out, int out_size, void* d_ws, size_t ws_size,
                              hipStream_t stream) {
  (void)in_sizes; (void)n_in; (void)out_size; (void)ws_size;
  const float* x        = (const float*)d_in[0];
  const float* x_pair   = (const float*)d_in[1];
  const unsigned char* pad_mask = (const unsigned char*)d_in[2];
  const unsigned char* exp_mask = (const unsigned char*)d_in[3];
  const int* pos_ids    = (const int*)d_in[4];
  const int* oc         = (const int*)d_in[5];
  const float* wq = (const float*)d_in[6];
  const float* wk = (const float*)d_in[7];
  const float* wv = (const float*)d_in[8];
  const float* wo = (const float*)d_in[9];
  const float* fc1 = (const float*)d_in[10];
  const float* fc2 = (const float*)d_in[11];
  const float* top_g = (const float*)d_in[12];
  const float* top_b = (const float*)d_in[13];
  const float* mid_g = (const float*)d_in[14];
  const float* mid_b = (const float*)d_in[15];
  const float* pair_g = (const float*)d_in[16];
  const float* pair_b = (const float*)d_in[17];
  const float* pw1 = (const float*)d_in[18];
  const float* pw2 = (const float*)d_in[19];

  float* out_x    = (float*)d_out;                       // (L,B,D)
  float* out_pair = (float*)d_out + (size_t)LB * Dd;     // (L,LK,B,HP)

  // ---- workspace carve-out ------------------------------------------------
  char* ws = (char*)d_ws;
  size_t off = 0;
  auto alloc = [&](size_t bytes) -> void* {
    off = (off + 255) & ~(size_t)255;
    void* p = ws + off;
    off += bytes;
    return p;
  };
  _Float16* h16   = (_Float16*)alloc((size_t)LB * Dd * 2);
  _Float16* kv16  = (_Float16*)alloc((size_t)LKB * Dd * 2);
  _Float16* wq16  = (_Float16*)alloc((size_t)Dd * Dd * 2);
  _Float16* wk16  = (_Float16*)alloc((size_t)Dd * Dd * 2);
  _Float16* wv16  = (_Float16*)alloc((size_t)Dd * Dd * 2);
  _Float16* wo16  = (_Float16*)alloc((size_t)Dd * Dd * 2);
  _Float16* fc1T  = (_Float16*)alloc((size_t)FFd * Dd * 2);
  _Float16* fc2T  = (_Float16*)alloc((size_t)Dd * FFd * 2);
  _Float16* pw1T  = (_Float16*)alloc((size_t)64 * Dd * 2);
  _Float16* pw2T  = (_Float16*)alloc((size_t)32 * 64 * 2);
  float*    qf    = (float*)alloc((size_t)LB * Dd * 4);
  float*    kf    = (float*)alloc((size_t)LKB * Dd * 4);
  float*    vf    = (float*)alloc((size_t)LKB * Dd * 4);
  _Float16* q16   = (_Float16*)alloc((size_t)Bd * Hd * Ld * DHd * 2);
  _Float16* k16   = (_Float16*)alloc((size_t)Bd * Hd * LKd * DHd * 2);
  _Float16* vt16  = (_Float16*)alloc((size_t)Bd * Hd * DHd * LKd * 2);
  _Float16* attn16= (_Float16*)alloc((size_t)LB * Dd * 2);
  float*    x1    = (float*)alloc((size_t)LB * Dd * 4);
  _Float16* h2_16 = (_Float16*)alloc((size_t)LB * Dd * 2);
  _Float16* ff16  = (_Float16*)alloc((size_t)LB * FFd * 2);
  _Float16* hp16  = (_Float16*)alloc((size_t)LB * Dd * 2);
  _Float16* xpa16 = (_Float16*)alloc((size_t)LB * 64 * 2);
  float*    xpi   = (float*)alloc((size_t)LB * HPd * 4);
  float*    xpj   = (float*)alloc((size_t)LKB * HPd * 4);

  auto blocks = [](size_t n) { return (unsigned)((n + 255) / 256); };

  // 1) weights -> f16, transposed (Bt layout: N x K)
  cvt_wT_kernel<<<blocks((size_t)Dd * Dd), 256, 0, stream>>>(wq, wq16, Dd, Dd);
  cvt_wT_kernel<<<blocks((size_t)Dd * Dd), 256, 0, stream>>>(wk, wk16, Dd, Dd);
  cvt_wT_kernel<<<blocks((size_t)Dd * Dd), 256, 0, stream>>>(wv, wv16, Dd, Dd);
  cvt_wT_kernel<<<blocks((size_t)Dd * Dd), 256, 0, stream>>>(wo, wo16, Dd, Dd);
  cvt_wT_kernel<<<blocks((size_t)Dd * FFd), 256, 0, stream>>>(fc1, fc1T, Dd, FFd);
  cvt_wT_kernel<<<blocks((size_t)FFd * Dd), 256, 0, stream>>>(fc2, fc2T, FFd, Dd);
  cvt_wT_kernel<<<blocks((size_t)Dd * 64), 256, 0, stream>>>(pw1, pw1T, Dd, 64);
  cvt_wT_kernel<<<blocks((size_t)64 * 32), 256, 0, stream>>>(pw2, pw2T, 64, 32);

  // 2) top LN, kv gather
  ln_f16_kernel<<<LB, 256, 0, stream>>>(x, top_g, top_b, h16, Dd);
  kv_gather_kernel<<<blocks((size_t)LKB * Dd), 256, 0, stream>>>(h16, oc, kv16);

  // 3) Q/K/V projections (WMMA)
  {
    dim3 g(Dd / 128, LB / 64);
    gemm_wmma_kernel<<<g, 256, 0, stream>>>(h16, wq16, nullptr, qf, nullptr,
                                            LB, Dd, Dd, 0);
  }
  {
    dim3 g(Dd / 128, LKB / 64);
    gemm_wmma_kernel<<<g, 256, 0, stream>>>(kv16, wk16, nullptr, kf, nullptr,
                                            LKB, Dd, Dd, 0);
    gemm_wmma_kernel<<<g, 256, 0, stream>>>(kv16, wv16, nullptr, vf, nullptr,
                                            LKB, Dd, Dd, 0);
  }

  // 4) RoPE + layout shuffles for attention
  rope_q_kernel<<<blocks((size_t)Ld * Bd * Hd * 32), 256, 0, stream>>>(qf, pos_ids, q16);
  rope_k_kernel<<<blocks((size_t)LKd * Bd * Hd * 32), 256, 0, stream>>>(kf, pos_ids, oc, k16);
  vtrans_kernel<<<blocks((size_t)LKB * Dd), 256, 0, stream>>>(vf, vt16);

  // 5) fused attention
  {
    dim3 g(Ld / 64, Bd * Hd);
    attn_kernel<<<g, 256, 0, stream>>>(q16, k16, vt16, pad_mask, exp_mask, attn16);
  }

  // 6) output projection + residual -> x1
  {
    dim3 g(Dd / 128, LB / 64);
    gemm_wmma_kernel<<<g, 256, 0, stream>>>(attn16, wo16, x, x1, nullptr,
                                            LB, Dd, Dd, 0);
  }

  // 7) FFN: mid LN -> fc1 (GELU) -> fc2 (+residual) -> out_x
  ln_f16_kernel<<<LB, 256, 0, stream>>>(x1, mid_g, mid_b, h2_16, Dd);
  {
    dim3 g(FFd / 128, LB / 64);
    gemm_wmma_kernel<<<g, 256, 0, stream>>>(h2_16, fc1T, nullptr, nullptr, ff16,
                                            LB, FFd, Dd, 1);
  }
  {
    dim3 g(Dd / 128, LB / 64);
    gemm_wmma_kernel<<<g, 256, 0, stream>>>(ff16, fc2T, x1, out_x, nullptr,
                                            LB, Dd, FFd, 0);
  }

  // 8) pair head: pair LN -> pw1 (SiLU) -> pw2 -> x_p_i
  ln_f16_kernel<<<LB, 256, 0, stream>>>(out_x, pair_g, pair_b, hp16, Dd);
  {
    dim3 g(1, LB / 64);   // N = 64: waves tn=2,3 early-out
    gemm_wmma_kernel<<<g, 256, 0, stream>>>(hp16, pw1T, nullptr, nullptr, xpa16,
                                            LB, 64, Dd, 2);
  }
  {
    dim3 g(1, LB / 64);   // N = 32: only tn=0 waves active
    gemm_wmma_kernel<<<g, 256, 0, stream>>>(xpa16, pw2T, nullptr, xpi, nullptr,
                                            LB, HPd, 64, 0);
  }

  // 9) x_p_j gather + streaming pair outer-product update
  xpj_kernel<<<blocks((size_t)LKd * Bd * HPd), 256, 0, stream>>>(xpi, oc, xpj);
  {
    const size_t n = (size_t)Ld * LKd * Bd * HPd;
    pair_outer_kernel<<<blocks(n), 256, 0, stream>>>(x_pair, xpi, xpj, out_pair, n);
  }
}
